// WindowAttention_85839216378491
// MI455X (gfx1250) — matrix operand order repaired and checked by reading, not compiled, not run on hardware
//
#include <hip/hip_runtime.h>
#include <hip/hip_bf16.h>
#include <cstdint>
#include <cstddef>

#define EMBED 1024
#define NHEAD 16
#define HDIM  64
#define WSZ   256
#define NWIN  64
#define MTOK  (NWIN * WSZ)   /* 16384 */
#define LTOK  (MTOK - 1)     /* 16383 */

typedef __attribute__((ext_vector_type(8)))  float  v8f;
typedef __attribute__((ext_vector_type(16))) __bf16 v16bf;

union AV  { v16bf v; uint4 q[2]; };
union BF8 { uint4 u; __bf16 e[8]; };

__device__ inline v8f wmma_bf16(v16bf a, v16bf b, v8f c) {
  return __builtin_amdgcn_wmma_f32_16x16x32_bf16(false, a, false, b, (short)0, c,
                                                 false, false);
}

__device__ inline void load8bf(float* dst, const __bf16* src) {
  BF8 t; t.u = *(const uint4*)src;
#pragma unroll
  for (int i = 0; i < 8; ++i) dst[i] = (float)t.e[i];
}
__device__ inline void load8f(float* dst, const float* src) {
  float4 a = *(const float4*)src;
  float4 b = *(const float4*)(src + 4);
  dst[0]=a.x; dst[1]=a.y; dst[2]=a.z; dst[3]=a.w;
  dst[4]=b.x; dst[5]=b.y; dst[6]=b.z; dst[7]=b.w;
}

// Async global->LDS 32-byte copy (two B128 transfers, GVS mode).
// ldsaddr: wave-relative LDS byte address (generic-pointer low 32 bits),
// gbyteoff: per-lane byte offset added to the uniform SGPR base pointer.
__device__ inline void async_copy32(unsigned ldsaddr, const void* base,
                                    unsigned gbyteoff) {
  asm volatile("global_load_async_to_lds_b128 %0, %1, %2"
               :: "v"(ldsaddr), "v"(gbyteoff), "s"(base) : "memory");
  asm volatile("global_load_async_to_lds_b128 %0, %1, %2 offset:16"
               :: "v"(ldsaddr), "v"(gbyteoff), "s"(base) : "memory");
}

// ---------------------------------------------------------------- prep ------
__global__ void __launch_bounds__(256)
cvt_f32_bf16(const float* __restrict__ src, __bf16* __restrict__ dst, size_t n) {
  size_t i = (size_t)blockIdx.x * blockDim.x + threadIdx.x;
  size_t stride = (size_t)gridDim.x * blockDim.x;
  for (; i < n; i += stride) dst[i] = (__bf16)src[i];
}

__global__ void __launch_bounds__(256)
build_xp_bf16(const float* __restrict__ x, __bf16* __restrict__ dst,
              size_t nx, size_t ntot) {
  size_t i = (size_t)blockIdx.x * blockDim.x + threadIdx.x;
  size_t stride = (size_t)gridDim.x * blockDim.x;
  for (; i < ntot; i += stride) dst[i] = (i < nx) ? (__bf16)x[i] : (__bf16)0.0f;
}

// ------------------------------------------------- GEMM  C = A * W^T + b ----
// A: [Mrows][1024] bf16 row-major, W: [N][1024] bf16 row-major (so B^T tile is
// just a row-major slab of W). Block = 256 thr (8 waves, 4x2), tile 128x64.
// Tiles are staged with async global->LDS copies (ASYNCcnt).
template <bool OUT_BF16>
__global__ void __launch_bounds__(256)
gemm_xwt_bf16(const __bf16* __restrict__ A, const __bf16* __restrict__ W,
              const float* __restrict__ bias, void* __restrict__ out,
              int N, int rowLimit) {
  __shared__ __bf16 sA[128 * 32];
  __shared__ __bf16 sB[64 * 32];

  const int t    = threadIdx.x;
  const int lane = t & 31;
  const int wv   = t >> 5;
  const int nl   = lane & 15;
  const int hi   = lane >> 4;
  const int wr   = wv >> 1;   // 0..3
  const int wc   = wv & 1;    // 0..1
  const int m0   = blockIdx.y * 128;
  const int n0   = blockIdx.x * 64;

  const int arow = t >> 1, asc = (t & 1) * 16;   // sA loader: 128 rows x 2 segs
  const int brow = t >> 1, bsc = (t & 1) * 16;   // sB loader (t < 128)

  const unsigned ldsA = (unsigned)(size_t)&sA[arow * 32 + asc];
  const unsigned ldsB = (unsigned)(size_t)&sB[brow * 32 + bsc];

  v8f acc[2][2];
#pragma unroll
  for (int ai = 0; ai < 2; ++ai)
#pragma unroll
    for (int bi = 0; bi < 2; ++bi)
      acc[ai][bi] = v8f{0.f,0.f,0.f,0.f,0.f,0.f,0.f,0.f};

  for (int k0 = 0; k0 < 1024; k0 += 32) {
    __syncthreads();
    {
      const unsigned ga = (unsigned)(((size_t)(m0 + arow) * 1024 + k0 + asc) * 2);
      async_copy32(ldsA, A, ga);
      if (k0 + 32 < 1024)
        __builtin_prefetch(A + (size_t)(m0 + arow) * 1024 + k0 + 32 + asc, 0, 0);
      if (t < 128) {
        const unsigned gb =
            (unsigned)(((size_t)(n0 + brow) * 1024 + k0 + bsc) * 2);
        async_copy32(ldsB, W, gb);
      }
    }
    asm volatile("s_wait_asynccnt 0x0" ::: "memory");
    __syncthreads();

#pragma unroll
    for (int ai = 0; ai < 2; ++ai) {
      AV a;
      const int am = wr * 32 + ai * 16 + nl;
      a.q[0] = *(const uint4*)&sA[am * 32 + hi * 8];
      a.q[1] = *(const uint4*)&sA[am * 32 + hi * 8 + 16];
#pragma unroll
      for (int bi = 0; bi < 2; ++bi) {
        AV b;
        const int bn = wc * 32 + bi * 16 + nl;
        b.q[0] = *(const uint4*)&sB[bn * 32 + hi * 8];
        b.q[1] = *(const uint4*)&sB[bn * 32 + hi * 8 + 16];
        acc[ai][bi] = wmma_bf16(a.v, b.v, acc[ai][bi]);
      }
    }
  }

#pragma unroll
  for (int ai = 0; ai < 2; ++ai)
#pragma unroll
    for (int bi = 0; bi < 2; ++bi) {
      const int n = n0 + wc * 32 + bi * 16 + nl;
      const float bv = bias[n];
#pragma unroll
      for (int rr = 0; rr < 8; ++rr) {
        const int m = m0 + wr * 32 + ai * 16 + rr + 8 * hi;
        if (m < rowLimit) {
          const float val = acc[ai][bi][rr] + bv;
          if (OUT_BF16) ((__bf16*)out)[(size_t)m * N + n] = (__bf16)val;
          else          ((float*)out)[(size_t)m * N + n]  = val;
        }
      }
    }
}

// ------------------------------------------- fused window attention --------
// One block per (window, head). Gather + RoPE fused into load phase.
__global__ void __launch_bounds__(256)
window_attn(const __bf16* __restrict__ qkv, const float* __restrict__ cosp,
            const float* __restrict__ sinp, const int* __restrict__ kvidx,
            __bf16* __restrict__ ori) {
  __shared__ __bf16 ldsQ[WSZ * HDIM];   // [j][d] row-major, RoPE applied
  __shared__ __bf16 ldsK[WSZ * HDIM];   // [j][d]
  __shared__ __bf16 ldsVT[HDIM * WSZ];  // [d][j]  (transposed V)
  __shared__ __bf16 pst[8 * 16 * 32];   // per-wave P relayout staging

  const int w = blockIdx.x, h = blockIdx.y;
  const int t = threadIdx.x;              // token row j in window
  const int p = w * WSZ + t;              // global window-ordered position
  const int r = kvidx[p];                 // source token row

  const __bf16* qr = qkv + (size_t)r * (3 * EMBED) + h * HDIM;         // q
  const __bf16* kr = qr + EMBED;                                       // k
  const __bf16* vr = qr + 2 * EMBED;                                   // v
  const float*  cb = cosp + (size_t)p * HDIM;
  const float*  sb = sinp + (size_t)p * HDIM;

#pragma unroll
  for (int c = 0; c < 4; ++c) {
    float clo[8], slo[8], chi[8], shi[8];
    load8f(clo, cb + c * 8);       load8f(slo, sb + c * 8);
    load8f(chi, cb + c * 8 + 32);  load8f(shi, sb + c * 8 + 32);
    float lo[8], hv[8];
    load8bf(lo, qr + c * 8); load8bf(hv, qr + c * 8 + 32);
#pragma unroll
    for (int i = 0; i < 8; ++i) {
      ldsQ[t * HDIM + c * 8 + i]      = (__bf16)(lo[i] * clo[i] - hv[i] * slo[i]);
      ldsQ[t * HDIM + c * 8 + 32 + i] = (__bf16)(hv[i] * chi[i] + lo[i] * shi[i]);
    }
    load8bf(lo, kr + c * 8); load8bf(hv, kr + c * 8 + 32);
#pragma unroll
    for (int i = 0; i < 8; ++i) {
      ldsK[t * HDIM + c * 8 + i]      = (__bf16)(lo[i] * clo[i] - hv[i] * slo[i]);
      ldsK[t * HDIM + c * 8 + 32 + i] = (__bf16)(hv[i] * chi[i] + lo[i] * shi[i]);
    }
  }
#pragma unroll
  for (int c = 0; c < 8; ++c) {
    float vv[8];
    load8bf(vv, vr + c * 8);
#pragma unroll
    for (int i = 0; i < 8; ++i)
      ldsVT[(c * 8 + i) * WSZ + t] = (__bf16)vv[i];
  }
  __syncthreads();

  const int lane = t & 31, wv = t >> 5;
  const int nl = lane & 15, hi = lane >> 4;

  for (int mi = 0; mi < 2; ++mi) {
    const int m0 = wv * 32 + mi * 16;     // 16-row strip of Q

    // ---- S = scale * Q K^T  (16 x 256 strip in registers) ----
    v8f sacc[16];
#pragma unroll
    for (int nt = 0; nt < 16; ++nt)
      sacc[nt] = v8f{0.f,0.f,0.f,0.f,0.f,0.f,0.f,0.f};

#pragma unroll
    for (int kk = 0; kk < HDIM; kk += 32) {
      AV a;
      a.q[0] = *(const uint4*)&ldsQ[(m0 + nl) * HDIM + kk + hi * 8];
      a.q[1] = *(const uint4*)&ldsQ[(m0 + nl) * HDIM + kk + hi * 8 + 16];
#pragma unroll
      for (int nt = 0; nt < 16; ++nt) {
        AV b;
        const int base = (nt * 16 + nl) * HDIM + kk + hi * 8;
        b.q[0] = *(const uint4*)&ldsK[base];
        b.q[1] = *(const uint4*)&ldsK[base + 16];
        sacc[nt] = wmma_bf16(a.v, b.v, sacc[nt]);
      }
    }

    // ---- row softmax (rows live in 16-lane halves) ----
    const float scale = 0.125f;  // HDIM^-0.5
#pragma unroll
    for (int nt = 0; nt < 16; ++nt) sacc[nt] *= scale;

    float rinv[8];
#pragma unroll
    for (int rr = 0; rr < 8; ++rr) {
      float mx = -3.0e38f;
#pragma unroll
      for (int nt = 0; nt < 16; ++nt) mx = fmaxf(mx, sacc[nt][rr]);
#pragma unroll
      for (int off = 1; off < 16; off <<= 1)
        mx = fmaxf(mx, __shfl_xor(mx, off, 32));
      float sum = 0.f;
#pragma unroll
      for (int nt = 0; nt < 16; ++nt) {
        const float e = __expf(sacc[nt][rr] - mx);
        sacc[nt][rr] = e;
        sum += e;
      }
#pragma unroll
      for (int off = 1; off < 16; off <<= 1)
        sum += __shfl_xor(sum, off, 32);
      rinv[rr] = 1.0f / sum;
    }
#pragma unroll
    for (int nt = 0; nt < 16; ++nt)
#pragma unroll
      for (int rr = 0; rr < 8; ++rr) sacc[nt][rr] *= rinv[rr];

    // ---- O = P V  (P relayout through wave-private LDS staging) ----
    v8f oacc[4];
#pragma unroll
    for (int dt = 0; dt < 4; ++dt)
      oacc[dt] = v8f{0.f,0.f,0.f,0.f,0.f,0.f,0.f,0.f};

#pragma unroll
    for (int jc = 0; jc < 8; ++jc) {          // 32 K-columns of P per step
#pragma unroll
      for (int tt = 0; tt < 2; ++tt) {
        const int nt = jc * 2 + tt;
#pragma unroll
        for (int rr = 0; rr < 8; ++rr)
          pst[wv * 512 + (rr + 8 * hi) * 32 + tt * 16 + nl] =
              (__bf16)sacc[nt][rr];
      }
      asm volatile("s_wait_dscnt 0" ::: "memory");  // cross-lane LDS RAW
      AV a;
      a.q[0] = *(const uint4*)&pst[wv * 512 + nl * 32 + hi * 8];
      a.q[1] = *(const uint4*)&pst[wv * 512 + nl * 32 + hi * 8 + 16];
#pragma unroll
      for (int dt = 0; dt < 4; ++dt) {
        AV b;
        const int base = (dt * 16 + nl) * WSZ + jc * 32 + hi * 8;
        b.q[0] = *(const uint4*)&ldsVT[base];
        b.q[1] = *(const uint4*)&ldsVT[base + 16];
        oacc[dt] = wmma_bf16(a.v, b.v, oacc[dt]);
      }
    }

    // ---- scatter O through kv_indices into ori ----
#pragma unroll
    for (int rr = 0; rr < 8; ++rr) {
      const int pgl  = w * WSZ + m0 + rr + 8 * hi;
      const int dest = kvidx[pgl];
      __bf16* orow = ori + (size_t)dest * EMBED + h * HDIM;
#pragma unroll
      for (int dt = 0; dt < 4; ++dt)
        orow[dt * 16 + nl] = (__bf16)oacc[dt][rr];
    }
  }
}

// ---------------------------------------------------------------- launch ----
extern "C" void kernel_launch(void* const* d_in, const int* in_sizes, int n_in,
                              void* d_out, int out_size, void* d_ws,
                              size_t ws_size, hipStream_t stream) {
  const float* x       = (const float*)d_in[0];
  const float* qkv_w   = (const float*)d_in[1];
  const float* qkv_b   = (const float*)d_in[2];
  const float* proj_w  = (const float*)d_in[3];
  const float* proj_b  = (const float*)d_in[4];
  const float* ropeCos = (const float*)d_in[6];
  const float* ropeSin = (const float*)d_in[7];
  const int*   kvIdx   = (const int*)d_in[8];
  (void)in_sizes; (void)n_in; (void)out_size; (void)ws_size;

  char* ws = (char*)d_ws;
  __bf16* xpB   = (__bf16*)(ws);                                 // 33,554,432 B
  __bf16* qkvwB = (__bf16*)(ws + (size_t)33554432);              //  6,291,456 B
  __bf16* pwB   = (__bf16*)(ws + (size_t)39845888);              //  2,097,152 B
  __bf16* qkvB  = (__bf16*)(ws + (size_t)41943040);              // 100,663,296 B
  __bf16* oriB  = (__bf16*)(ws + (size_t)142606336);             // 33,554,432 B

  // Stage 0: bf16 conversions (+ zero padding row for xp)
  build_xp_bf16<<<8192, 256, 0, stream>>>(x, xpB,
      (size_t)LTOK * EMBED, (size_t)MTOK * EMBED);
  cvt_f32_bf16<<<4096, 256, 0, stream>>>(qkv_w, qkvwB, (size_t)3 * EMBED * EMBED);
  cvt_f32_bf16<<<2048, 256, 0, stream>>>(proj_w, pwB, (size_t)EMBED * EMBED);

  // Stage 1: QKV = xp @ qkv_w^T + qkv_b   (bf16 out)
  gemm_xwt_bf16<true><<<dim3((3 * EMBED) / 64, MTOK / 128), 256, 0, stream>>>(
      xpB, qkvwB, qkv_b, qkvB, 3 * EMBED, MTOK);

  // Stage 2: per-(window, head) attention with fused gather + RoPE + scatter
  window_attn<<<dim3(NWIN, NHEAD), 256, 0, stream>>>(
      qkvB, ropeCos, ropeSin, kvIdx, oriB);

  // Stage 3: out[:L] = ori @ proj_w^T + proj_b  (fp32 out, rows >= L dropped)
  gemm_xwt_bf16<false><<<dim3(EMBED / 64, MTOK / 128), 256, 0, stream>>>(
      oriB, pwB, proj_b, d_out, EMBED, LTOK);
}
